// MultiHeadAttention_17239998726558
// MI455X (gfx1250) — matrix operand order repaired and checked by reading, not compiled, not run on hardware
//
#include <hip/hip_runtime.h>

// ---------------------------------------------------------------------------
// MHA forward for B=2,S=2048,D=1024,H=16,HD=64 on gfx1250 (wave32, WMMA).
// Pipeline: cvt f32->bf16, LDS-staged double-buffered WMMA GEMMs (QKV proj,
// K stored transposed per head), flash attention with WMMA score/ctx matmuls,
// WMMA output projection + bias.
// ---------------------------------------------------------------------------

typedef __attribute__((ext_vector_type(16))) __bf16     bf16x16;
typedef __attribute__((ext_vector_type(4)))  __bf16     bf16x4;
typedef __attribute__((ext_vector_type(8)))  float      f32x8;
typedef __attribute__((ext_vector_type(4)))  unsigned   u32x4;

union FragAB { bf16x16 v; u32x4 q[2]; };

#define WMMA_BF16(a, b, c) \
  __builtin_amdgcn_wmma_f32_16x16x32_bf16(false, (a), false, (b), (short)0, (c), false, false)

constexpr int Bb = 2, Ss = 2048, Dd = 1024, Hh = 16, HD = 64;
constexpr int GM = Bb * Ss;   // 4096 rows for all big GEMMs
constexpr int GN = Dd;        // 1024
constexpr int GK = Dd;        // 1024

constexpr int MODE_QV  = 0;   // bf16 out, [b,h,s,hd]
constexpr int MODE_KT  = 1;   // bf16 out, [b,h,hd,s]  (transposed K)
constexpr int MODE_OUT = 2;   // f32 out + bias, [m,n]

constexpr int LDA = 40;       // A LDS row stride (bf16), 80B: conflict-free frag reads
constexpr int LDB = 136;      // B LDS row stride (bf16), 272B: conflict-free frag reads

// ---------------------------------------------------------------------------
__global__ __launch_bounds__(256)
void cvt_f32_bf16(const float* __restrict__ src, __bf16* __restrict__ dst, int n) {
  int i = (blockIdx.x * blockDim.x + threadIdx.x) * 4;
  if (i + 3 < n) {
    float4 f = *(const float4*)(src + i);
    bf16x4 o;
    o.x = (__bf16)f.x; o.y = (__bf16)f.y; o.z = (__bf16)f.z; o.w = (__bf16)f.w;
    *(bf16x4*)(dst + i) = o;
  }
}

// ---------------------------------------------------------------------------
// C[M,N] = A[M,K] * W[K,N] (bf16 in, f32 acc).
// Block = 256 threads (8 waves), 128x128 block tile, BK=32, double-buffered
// LDS staging. Waves arranged 2(M) x 4(N); each wave computes a 64x32 tile
// with 4x2 WMMA accumulators (8 WMMAs per K-step).
template <int MODE>
__global__ __launch_bounds__(256)
void gemm_bf16(const __bf16* __restrict__ A, const __bf16* __restrict__ W,
               void* __restrict__ out, const float* __restrict__ bias) {
  __shared__ __align__(16) __bf16 Asm[2][128 * LDA];   // 2 x 10240 B
  __shared__ __align__(16) __bf16 Bsm[2][32 * LDB];    // 2 x  8704 B

  const int tid    = threadIdx.x;
  const int lane   = tid & 31;
  const int wave   = tid >> 5;
  const int half   = lane >> 4;
  const int mrow   = lane & 15;
  const int krow   = mrow + half * 16;
  const int wave_m = wave & 1;       // 0..1 -> 64-row group
  const int wave_n = wave >> 1;      // 0..3 -> 32-col group
  const int m0 = blockIdx.y * 128;
  const int n0 = blockIdx.x * 128;

  // staging assignment: each thread moves 32B of A and 32B of B per K-step
  const int row_a = tid >> 1, seg_a = (tid & 1) * 16;   // A: 128 rows x 32 k
  const int row_b = tid >> 3, seg_b = (tid & 7) * 16;   // B:  32 rows x 128 n
  const __bf16* Ag = A + (m0 + row_a) * GK + seg_a;
  const __bf16* Bg = W + row_b * GN + n0 + seg_b;

  // prologue: stage K-tile 0 into buffer 0
  u32x4 ra0 = *(const u32x4*)(Ag);
  u32x4 ra1 = *(const u32x4*)(Ag + 8);
  u32x4 rb0 = *(const u32x4*)(Bg);
  u32x4 rb1 = *(const u32x4*)(Bg + 8);
  {
    __bf16* as = &Asm[0][row_a * LDA + seg_a];
    *(u32x4*)as = ra0; *(u32x4*)(as + 8) = ra1;
    __bf16* bs = &Bsm[0][row_b * LDB + seg_b];
    *(u32x4*)bs = rb0; *(u32x4*)(bs + 8) = rb1;
  }
  __syncthreads();

  f32x8 acc[4][2] = {};
  constexpr int NT = GK / 32;        // 32 K-tiles
  for (int kt = 0; kt < NT; ++kt) {
    const int cur = kt & 1;
    // prefetch next K-tile into registers (overlaps with WMMA compute below)
    if (kt + 1 < NT) {
      const int k = (kt + 1) * 32;
      ra0 = *(const u32x4*)(Ag + k);
      ra1 = *(const u32x4*)(Ag + k + 8);
      rb0 = *(const u32x4*)(Bg + (size_t)k * GN);
      rb1 = *(const u32x4*)(Bg + (size_t)k * GN + 8);
      if (kt + 2 < NT) {   // warm GL2 for the tile after (global_prefetch_b8)
        __builtin_prefetch(Ag + (kt + 2) * 32, 0, 1);
        __builtin_prefetch(Bg + (size_t)(kt + 2) * 32 * GN, 0, 1);
      }
    }

    // compute from buf[cur]
    FragAB af[4], bfr[2];
    #pragma unroll
    for (int i = 0; i < 4; ++i) {
      const __bf16* ar = &Asm[cur][(wave_m * 64 + i * 16 + mrow) * LDA];
      af[i].q[0] = *(const u32x4*)(ar + half * 8);
      af[i].q[1] = *(const u32x4*)(ar + 16 + half * 8);
    }
    #pragma unroll
    for (int j = 0; j < 2; ++j) {
      const __bf16* br = &Bsm[cur][wave_n * 32 + j * 16 + mrow];
      #pragma unroll
      for (int s = 0; s < 16; ++s)
        bfr[j].v[s] = br[(s < 8 ? 8 * half + s : 8 + 8 * half + s) * LDB];
    }
    #pragma unroll
    for (int i = 0; i < 4; ++i)
      #pragma unroll
      for (int j = 0; j < 2; ++j)
        acc[i][j] = WMMA_BF16(af[i].v, bfr[j].v, acc[i][j]);

    // store the prefetched tile into the other buffer
    if (kt + 1 < NT) {
      __syncthreads();
      const int nxt = cur ^ 1;
      __bf16* as = &Asm[nxt][row_a * LDA + seg_a];
      *(u32x4*)as = ra0; *(u32x4*)(as + 8) = ra1;
      __bf16* bs = &Bsm[nxt][row_b * LDB + seg_b];
      *(u32x4*)bs = rb0; *(u32x4*)(bs + 8) = rb1;
      __syncthreads();
    }
  }

  // C/D layout: vgpr r, lane -> (M = r + 8*half, N = mrow)
  #pragma unroll
  for (int i = 0; i < 4; ++i)
    #pragma unroll
    for (int j = 0; j < 2; ++j)
      #pragma unroll
      for (int r = 0; r < 8; ++r) {
        const int m = m0 + wave_m * 64 + i * 16 + r + 8 * half;
        const int n = n0 + wave_n * 32 + j * 16 + mrow;
        const float v = acc[i][j][r];
        if (MODE == MODE_OUT) {
          ((float*)out)[(size_t)m * GN + n] = v + bias[n];
        } else {
          const int bbx = m >> 11, s = m & (Ss - 1);
          const int h   = n >> 6,  hd = n & (HD - 1);
          const int idx = (MODE == MODE_QV)
                        ? (((bbx * Hh + h) * Ss + s) * HD + hd)    // [b,h,s,hd]
                        : (((bbx * Hh + h) * HD + hd) * Ss + s);   // [b,h,hd,s]
          ((__bf16*)out)[idx] = (__bf16)v;
        }
      }
}

// ---------------------------------------------------------------------------
// Flash attention. grid = (B*H, S/64), block = 128 (4 waves). Each wave owns
// 16 q-rows, loops causal k-blocks of 32, keeps a 16x64 f32 ctx accumulator.
__global__ __launch_bounds__(128)
void attention_kernel(const __bf16* __restrict__ Q,   // [b,h,s,hd]
                      const __bf16* __restrict__ Kt,  // [b,h,hd,s]
                      const __bf16* __restrict__ V,   // [b,h,s,hd]
                      __bf16* __restrict__ ctx)       // [b,s,h*hd]
{
  __shared__ __align__(16) __bf16 lds[4][16 * 40];  // per-wave 16x32 P tile
  const int lane = threadIdx.x & 31;
  const int wave = threadIdx.x >> 5;
  const int half = lane >> 4;
  const int mrow = lane & 15;
  const int krow = mrow + half * 16;
  const int bh   = blockIdx.x;                 // b*H + h
  const int q0   = blockIdx.y * 64 + wave * 16;
  const int bb   = bh >> 4;
  const int h    = bh & (Hh - 1);

  const __bf16* Qh = Q  + bh * (Ss * HD);
  const __bf16* Kh = Kt + bh * (HD * Ss);
  const __bf16* Vh = V  + bh * (Ss * HD);
  __bf16* myLds = lds[wave];

  // Q A-frags, hd chunks [0,32) and [32,64): row M = mrow
  FragAB aq[2];
  {
    const __bf16* qrow = Qh + (q0 + mrow) * HD;
    aq[0].q[0] = *(const u32x4*)(qrow + half * 8);
    aq[0].q[1] = *(const u32x4*)(qrow + 16 + half * 8);
    aq[1].q[0] = *(const u32x4*)(qrow + 32 + half * 8);
    aq[1].q[1] = *(const u32x4*)(qrow + 48 + half * 8);
  }

  float m_i[8], l_i[8];
  #pragma unroll
  for (int r = 0; r < 8; ++r) { m_i[r] = -1e30f; l_i[r] = 0.0f; }
  f32x8 acc[4] = {};

  const int kend = q0 + 16;                     // causal bound for this wave
  for (int k0 = 0; k0 < kend; k0 += 32) {
    // ---- scores: two 16x16 tiles (cols k0..k0+15, k0+16..k0+31) ----
    FragAB b00, b01, b10, b11;                  // [tile][hd-chunk]
    const __bf16* kb0 = Kh + k0 + mrow;               // hd-chunk 0, key column
    const __bf16* kb1 = Kh + 32 * Ss + k0 + mrow;     // hd-chunk 1, key column
    #pragma unroll
    for (int s = 0; s < 16; ++s) {
      const int kk = (s < 8 ? 8 * half + s : 8 + 8 * half + s);
      b00.v[s] = kb0[kk * Ss];
      b10.v[s] = kb0[kk * Ss + 16];
      b01.v[s] = kb1[kk * Ss];
      b11.v[s] = kb1[kk * Ss + 16];
    }

    f32x8 s0 = {}, s1 = {};
    s0 = WMMA_BF16(aq[0].v, b00.v, s0);
    s0 = WMMA_BF16(aq[1].v, b01.v, s0);
    s1 = WMMA_BF16(aq[0].v, b10.v, s1);
    s1 = WMMA_BF16(aq[1].v, b11.v, s1);

    // ---- online softmax (rows live across 16 lanes of a half-wave) ----
    #pragma unroll
    for (int r = 0; r < 8; ++r) {
      const int row = q0 + r + 8 * half;
      float v0 = s0[r] * 0.125f;                // 1/sqrt(64)
      float v1 = s1[r] * 0.125f;
      if (k0 + mrow > row)      v0 = -1e30f;    // causal mask
      if (k0 + 16 + mrow > row) v1 = -1e30f;
      float mx = fmaxf(v0, v1);
      mx = fmaxf(mx, __shfl_xor(mx, 1));
      mx = fmaxf(mx, __shfl_xor(mx, 2));
      mx = fmaxf(mx, __shfl_xor(mx, 4));
      mx = fmaxf(mx, __shfl_xor(mx, 8));
      const float mn   = fmaxf(m_i[r], mx);
      const float corr = __expf(m_i[r] - mn);
      m_i[r] = mn;
      const float p0 = __expf(v0 - mn);
      const float p1 = __expf(v1 - mn);
      float rs = p0 + p1;
      rs += __shfl_xor(rs, 1);
      rs += __shfl_xor(rs, 2);
      rs += __shfl_xor(rs, 4);
      rs += __shfl_xor(rs, 8);
      l_i[r] = l_i[r] * corr + rs;
      #pragma unroll
      for (int t = 0; t < 4; ++t) acc[t][r] *= corr;
      // stage P into per-wave LDS tile (C-layout -> A-layout shuffle)
      myLds[(r + 8 * half) * 40 + mrow]      = (__bf16)p0;
      myLds[(r + 8 * half) * 40 + 16 + mrow] = (__bf16)p1;
    }
    asm volatile("s_wait_dscnt 0" ::: "memory");  // DS in-order per wave

    // P as A-frag (16x32 over k) from LDS
    FragAB pf;
    pf.q[0] = *(const u32x4*)(myLds + mrow * 40 + half * 8);
    pf.q[1] = *(const u32x4*)(myLds + mrow * 40 + 16 + half * 8);

    // ---- ctx += P @ V (4 tiles of 16 hd columns) ----
    #pragma unroll
    for (int t = 0; t < 4; ++t) {
      FragAB vf;
      const __bf16* vb = Vh + k0 * HD + t * 16 + mrow;
      #pragma unroll
      for (int s = 0; s < 16; ++s)
        vf.v[s] = vb[(s < 8 ? 8 * half + s : 8 + 8 * half + s) * HD];
      acc[t] = WMMA_BF16(pf.v, vf.v, acc[t]);
    }
  }

  // ---- normalize, write ctx as [b,s,h*hd] bf16 ----
  float inv[8];
  #pragma unroll
  for (int r = 0; r < 8; ++r) inv[r] = 1.0f / l_i[r];
  #pragma unroll
  for (int t = 0; t < 4; ++t)
    #pragma unroll
    for (int r = 0; r < 8; ++r) {
      const int s   = q0 + r + 8 * half;
      const int col = h * HD + t * 16 + mrow;
      ctx[(bb * Ss + s) * Dd + col] = (__bf16)(acc[t][r] * inv[r]);
    }
}

// ---------------------------------------------------------------------------
extern "C" void kernel_launch(void* const* d_in, const int* in_sizes, int n_in,
                              void* d_out, int out_size, void* d_ws, size_t ws_size,
                              hipStream_t stream) {
  const float* x  = (const float*)d_in[0];
  const float* Wq = (const float*)d_in[1];
  const float* Wk = (const float*)d_in[2];
  const float* Wv = (const float*)d_in[3];
  const float* Wo = (const float*)d_in[4];
  const float* bo = (const float*)d_in[5];

  // Workspace layout (bytes): everything bf16, ~48 MB total.
  char* ws = (char*)d_ws;
  __bf16* xb  = (__bf16*)(ws);               // x bf16:  4194304 el, 8388608 B
  __bf16* wqb = (__bf16*)(ws +  8388608);    // Wq bf16: 1048576 el
  __bf16* wkb = (__bf16*)(ws + 10485760);
  __bf16* wvb = (__bf16*)(ws + 12582912);
  __bf16* wob = (__bf16*)(ws + 14680064);
  __bf16* Qb  = (__bf16*)(ws + 16777216);    // [b,h,s,hd] bf16, 8388608 B
  __bf16* Ktb = (__bf16*)(ws + 25165824);    // [b,h,hd,s] bf16
  __bf16* Vb  = (__bf16*)(ws + 33554432);    // [b,h,s,hd] bf16
  __bf16* ctx = (__bf16*)(ws + 41943040);    // [b,s,d]    bf16

  // 1) downcast activations + weights
  cvt_f32_bf16<<<4194304 / 1024, 256, 0, stream>>>(x,  xb,  4194304);
  cvt_f32_bf16<<<1048576 / 1024, 256, 0, stream>>>(Wq, wqb, 1048576);
  cvt_f32_bf16<<<1048576 / 1024, 256, 0, stream>>>(Wk, wkb, 1048576);
  cvt_f32_bf16<<<1048576 / 1024, 256, 0, stream>>>(Wv, wvb, 1048576);
  cvt_f32_bf16<<<1048576 / 1024, 256, 0, stream>>>(Wo, wob, 1048576);

  // 2) QKV projections (K stored transposed per head)
  dim3 gg(GN / 128, GM / 128);  // (8, 32)
  gemm_bf16<MODE_QV><<<gg, 256, 0, stream>>>(xb, wqb, (void*)Qb,  nullptr);
  gemm_bf16<MODE_KT><<<gg, 256, 0, stream>>>(xb, wkb, (void*)Ktb, nullptr);
  gemm_bf16<MODE_QV><<<gg, 256, 0, stream>>>(xb, wvb, (void*)Vb,  nullptr);

  // 3) causal flash attention
  attention_kernel<<<dim3(Bb * Hh, Ss / 64), 128, 0, stream>>>(Qb, Ktb, Vb, ctx);

  // 4) output projection + bias (f32 out)
  gemm_bf16<MODE_OUT><<<gg, 256, 0, stream>>>(ctx, wob, d_out, bo);
}
